// InferenceMultiHeadAttention_83004537962943
// MI455X (gfx1250) — matrix operand order repaired and checked by reading, not compiled, not run on hardware
//
#include <hip/hip_runtime.h>

// ---------------- types ----------------
typedef __attribute__((ext_vector_type(16))) __bf16     v16bf;
typedef __attribute__((ext_vector_type(8)))  float      v8f;
typedef __attribute__((ext_vector_type(8)))  unsigned   v8u;

// ---------------- constants ----------------
constexpr int kE       = 256;          // embed dim
constexpr int kH       = 4;            // heads
constexpr int kSeq     = 32768;
constexpr float kScale = 0.125f;       // 1/sqrt(64)

constexpr int kWaves   = 4;            // waves per block in k2
constexpr int kChunks  = 2;            // 32-row chunks per wave
constexpr int kNBlk    = kSeq / (kWaves * kChunks * 32);   // 128 blocks
constexpr int kNPart   = kNBlk;                            // one partial per block

// workspace layout (in float units; B-frags stored as dwords at offset 0)
constexpr int kWsBOff    = 0;      // 8 chunks * 32 lanes * 8 dwords = 2048 dwords
constexpr int kWsCOff    = 2048;   // 16 floats (per-head score bias, padded)
constexpr int kPartOff   = 4096;   // partials
constexpr int kPartStride = 8 + kH * kE;   // m[4], l[4], t[4][256] = 1032 -> pad
constexpr int kPartPad   = 1040;

// ---------------- helpers ----------------
__device__ __forceinline__ unsigned f2bf(float f) {
    unsigned u = __float_as_uint(f);
    unsigned r = (u + 0x7FFFu + ((u >> 16) & 1u)) >> 16;   // round-to-nearest-even
    return r & 0xFFFFu;
}
__device__ __forceinline__ unsigned pack2(float lo, float hi) {
    return f2bf(lo) | (f2bf(hi) << 16);
}

// =====================================================================
// Kernel 1: q projection + effective score vectors, pre-packed bf16
// B-fragments in WMMA lane layout.
// =====================================================================
__global__ __launch_bounds__(256) void k1_prep(const float* __restrict__ x,
                                               const float* __restrict__ W,   // in_proj_weight (768x256)
                                               const float* __restrict__ b,   // in_proj_bias (768)
                                               float* __restrict__ ws) {
    __shared__ float sx[kE];
    __shared__ float sq[kE];
    __shared__ float sw[kH * kE];   // wq_eff (scale folded in)

    const int tid = threadIdx.x;
    sx[tid] = x[tid];
    __syncthreads();

    // q[j] = bq[j] + x[0,:] . Wq[j,:]
    {
        float acc = b[tid];
        const float* wr = W + (size_t)tid * kE;
        #pragma unroll 4
        for (int e = 0; e < kE; ++e) acc += sx[e] * wr[e];
        sq[tid] = acc;
    }
    __syncthreads();

    // wq_eff[h][e] = scale * sum_d q[h*64+d] * Wk[h*64+d][e]   (Wk rows 256..511)
    #pragma unroll
    for (int h = 0; h < kH; ++h) {
        float acc = 0.f;
        const int e = tid;
        #pragma unroll 4
        for (int d = 0; d < 64; ++d)
            acc += sq[h * 64 + d] * W[(size_t)(kE + h * 64 + d) * kE + e];
        sw[h * kE + e] = acc * kScale;
    }
    // c[h] = scale * bk_h . q_h  (pad heads 4..15 with zero)
    if (tid < 16) {
        float c = 0.f;
        if (tid < kH) {
            for (int d = 0; d < 64; ++d) c += b[kE + tid * 64 + d] * sq[tid * 64 + d];
            c *= kScale;
        }
        ws[kWsCOff + tid] = c;
    }
    __syncthreads();

    // Pack B fragments: for K-chunk c, lane l, dword j:
    //   lanes 0-15:  K = 32c + 2j, 2j+1   ; lanes 16-31: K = 32c + 16 + 2j, ...
    //   column n = l & 15 (head; zero for n >= 4)
    {
        const int c = tid >> 5, l = tid & 31;
        const int n = l & 15, g = (l >> 4) * 16;
        unsigned* wsb = (unsigned*)ws;   // dwords at offset 0
        #pragma unroll
        for (int j = 0; j < 8; ++j) {
            const int k0 = c * 32 + g + 2 * j;
            unsigned v = 0u;
            if (n < kH) v = pack2(sw[n * kE + k0], sw[n * kE + k0 + 1]);
            wsb[tid * 8 + j] = v;
        }
    }
}

// =====================================================================
// Kernel 2: fused streaming pass over x — WMMA scores + online softmax
// + WMMA weighted accumulation; one partial (m,l,t[4][256]) per block.
// =====================================================================
__global__ __launch_bounds__(kWaves * 32) void k2_stream(const float* __restrict__ x,
                                                         float* __restrict__ ws) {
    __shared__ __align__(16) unsigned short smem[kWaves * 32 * kE];   // 64 KB: x staging, reused for reduction

    const int tid  = threadIdx.x;
    const int lane = tid & 31;
    const int w    = tid >> 5;
    const int ln   = lane & 15;       // column / row-in-tile index
    const int hi   = lane >> 4;       // half-wave select

    const unsigned* wsb = (const unsigned*)ws;           // B-fragments
    const float     cn  = ws[kWsCOff + ln];              // per-head score bias (0 for padded heads)

    unsigned short* my = smem + w * 32 * kE;             // this wave's 32x256 bf16 tile

    const v8f zero8 = {0.f, 0.f, 0.f, 0.f, 0.f, 0.f, 0.f, 0.f};
    float m_run = -3.0e38f, l_run = 0.f;
    v8f accT[16];
    #pragma unroll
    for (int t = 0; t < 16; ++t) accT[t] = zero8;

    for (int ch = 0; ch < kChunks; ++ch) {
        const int    p = (blockIdx.x * kWaves + w) * kChunks + ch;
        const size_t R = (size_t)p * 32;

        // ---- stage 32x256 fp32 rows -> bf16 LDS (row-major) ----
        const float4* xr = (const float4*)(x + R * kE);
        #pragma unroll 4
        for (int r = 0; r < 32; ++r) {
            float4 a = xr[r * 64 + lane * 2];
            float4 c4 = xr[r * 64 + lane * 2 + 1];
            uint4 d;
            d.x = pack2(a.x, a.y);  d.y = pack2(a.z, a.w);
            d.z = pack2(c4.x, c4.y); d.w = pack2(c4.z, c4.w);
            *(uint4*)(my + r * kE + lane * 8) = d;
        }

        // ---- scores: two 16x16 WMMA tiles (rows 0-15, 16-31), K = 256 ----
        v8f s[2];
        #pragma unroll
        for (int T = 0; T < 2; ++T) {
            v8f acc = zero8;
            const unsigned short* row = my + (T * 16 + ln) * kE;
            #pragma unroll
            for (int c = 0; c < 8; ++c) {
                uint4 d0 = *(const uint4*)(row + c * 32 + hi * 16);
                uint4 d1 = *(const uint4*)(row + c * 32 + hi * 16 + 8);
                v8u au = {d0.x, d0.y, d0.z, d0.w, d1.x, d1.y, d1.z, d1.w};
                uint4 b0 = *(const uint4*)(wsb + (c * 32 + lane) * 8);
                uint4 b1 = *(const uint4*)(wsb + (c * 32 + lane) * 8 + 4);
                v8u bu = {b0.x, b0.y, b0.z, b0.w, b1.x, b1.y, b1.z, b1.w};
                acc = __builtin_amdgcn_wmma_f32_16x16x32_bf16(
                    false, __builtin_bit_cast(v16bf, au),
                    false, __builtin_bit_cast(v16bf, bu),
                    (short)0, acc, false, false);
            }
            s[T] = acc;
        }

        // ---- online softmax update (per head = lane%16; lane pairs share via xor-16) ----
        float mloc = m_run;
        #pragma unroll
        for (int i = 0; i < 8; ++i) {
            mloc = fmaxf(mloc, s[0][i] + cn);
            mloc = fmaxf(mloc, s[1][i] + cn);
        }
        const float mnew = fmaxf(mloc, __shfl_xor(mloc, 16, 32));
        const float rsc  = __expf(m_run - mnew);

        float p0[8], p1[8], lloc = 0.f;
        #pragma unroll
        for (int i = 0; i < 8; ++i) {
            p0[i] = __expf(s[0][i] + cn - mnew);  lloc += p0[i];
            p1[i] = __expf(s[1][i] + cn - mnew);  lloc += p1[i];
        }
        lloc += __shfl_xor(lloc, 16, 32);
        l_run = l_run * rsc + lloc;
        m_run = mnew;

        // p matrix is already in A-fragment layout (rows=heads) — no shuffle needed
        v8u pu = {pack2(p0[0], p0[1]), pack2(p0[2], p0[3]), pack2(p0[4], p0[5]), pack2(p0[6], p0[7]),
                  pack2(p1[0], p1[1]), pack2(p1[2], p1[3]), pack2(p1[4], p1[5]), pack2(p1[6], p1[7])};
        const v16bf Ap = __builtin_bit_cast(v16bf, pu);

        // ---- t[h, e] += p[h, row] * x[row, e] : 16 e-tiles, x read as B-frag from LDS ----
        #pragma unroll
        for (int t = 0; t < 16; ++t) {
            const unsigned short* colp = my + (hi * 8) * kE + t * 16 + ln;
            v8u xb;
            #pragma unroll
            for (int j = 0; j < 8; ++j)
                xb[j] = (unsigned)colp[(2 * j + (j >> 2) * 8) * kE] | ((unsigned)colp[(2 * j + (j >> 2) * 8 + 1) * kE] << 16);
            v8f cprev = accT[t];
            #pragma unroll
            for (int i = 0; i < 8; ++i) cprev[i] *= rsc;
            accT[t] = __builtin_amdgcn_wmma_f32_16x16x32_bf16(
                false, Ap, false, __builtin_bit_cast(v16bf, xb),
                (short)0, cprev, false, false);
        }
    }

    // ---- block-level reduction (reuse staging LDS as float scratch) ----
    __syncthreads();
    float* red = (float*)smem;
    // layout: [0, 4096) t per wave; [4096,4112) m; [4112,4128) l; [4128,4132) blk m; [4160,4176) scales
    if (lane < 16) {
        #pragma unroll
        for (int t = 0; t < 16; ++t)
            #pragma unroll
            for (int i = 0; i < kH; ++i)
                red[w * 1024 + i * kE + t * 16 + lane] = accT[t][i];
    }
    if (lane < kH) {
        red[4096 + w * 4 + lane] = m_run;
        red[4112 + w * 4 + lane] = l_run;
    }
    __syncthreads();
    if (tid < kH) {
        float mb = -3.0e38f;
        for (int w2 = 0; w2 < kWaves; ++w2) mb = fmaxf(mb, red[4096 + w2 * 4 + tid]);
        red[4128 + tid] = mb;
    }
    __syncthreads();
    if (tid < 16) {
        const int w2 = tid >> 2, h = tid & 3;
        red[4160 + tid] = __expf(red[4096 + w2 * 4 + h] - red[4128 + h]);
    }
    __syncthreads();

    const size_t PB = (size_t)kPartOff + (size_t)blockIdx.x * kPartPad;
    if (tid < kH) {
        float lb = 0.f;
        for (int w2 = 0; w2 < kWaves; ++w2)
            lb += red[4112 + w2 * 4 + tid] * red[4160 + w2 * 4 + tid];
        ws[PB + tid]      = red[4128 + tid];
        ws[PB + 4 + tid]  = lb;
    }
    #pragma unroll
    for (int k = 0; k < 8; ++k) {
        const int item = tid + k * (kWaves * 32);
        const int h = item >> 8, e = item & 255;
        float acc = 0.f;
        for (int w2 = 0; w2 < kWaves; ++w2)
            acc += red[w2 * 1024 + h * kE + e] * red[4160 + w2 * 4 + h];
        ws[PB + 8 + h * kE + e] = acc;
    }
}

// =====================================================================
// Kernel 3: combine partials, u = T/l, ctx = u.Wv^T + bv, out = ctx.Wo^T + bo
// =====================================================================
__global__ __launch_bounds__(256) void k3_final(const float* __restrict__ W,    // in_proj_weight
                                                const float* __restrict__ b,    // in_proj_bias
                                                const float* __restrict__ Wo,   // out_proj_weight
                                                const float* __restrict__ bo,   // out_proj_bias
                                                const float* __restrict__ ws,
                                                float* __restrict__ out) {
    __shared__ float sm[kH], sl[kH];
    __shared__ float ss[kNPart * kH];
    __shared__ float su[kH * kE];
    __shared__ float sctx[kE];

    const int tid = threadIdx.x;

    if (tid < kH) {
        float m = -3.0e38f;
        for (int p = 0; p < kNPart; ++p)
            m = fmaxf(m, ws[kPartOff + p * kPartPad + tid]);
        sm[tid] = m;
    }
    __syncthreads();
    for (int k = 0; k < (kNPart * kH + 255) / 256; ++k) {
        const int item = tid + k * 256;
        if (item < kNPart * kH) {
            const int p = item >> 2, h = item & 3;
            ss[p * 4 + h] = __expf(ws[kPartOff + p * kPartPad + h] - sm[h]);
        }
    }
    __syncthreads();
    if (tid < kH) {
        float L = 0.f;
        for (int p = 0; p < kNPart; ++p)
            L += ws[kPartOff + p * kPartPad + 4 + tid] * ss[p * 4 + tid];
        sl[tid] = L;
    }
    __syncthreads();

    // u[h][e] = (sum_p t_p[h][e] * scale_p) / L_h      (thread owns column e = tid)
    #pragma unroll
    for (int h = 0; h < kH; ++h) {
        float acc = 0.f;
        for (int p = 0; p < kNPart; ++p)
            acc += ws[kPartOff + p * kPartPad + 8 + h * kE + tid] * ss[p * 4 + h];
        su[h * kE + tid] = acc / sl[h];
    }
    __syncthreads();

    // ctx[j] = bv[j] + u[h] . Wv[j,:]   (Wv rows 512..767; j = tid, h = j/64)
    {
        const int h = tid >> 6;
        const float* wv = W + (size_t)(2 * kE + tid) * kE;
        float acc = b[2 * kE + tid];
        #pragma unroll 4
        for (int e = 0; e < kE; ++e) acc += su[h * kE + e] * wv[e];
        sctx[tid] = acc;
    }
    __syncthreads();

    // out[j] = bo[j] + ctx . Wo[j,:]
    {
        const float* wr = Wo + (size_t)tid * kE;
        float acc = bo[tid];
        #pragma unroll 4
        for (int i = 0; i < kE; ++i) acc += sctx[i] * wr[i];
        out[tid] = acc;
    }
}

// =====================================================================
extern "C" void kernel_launch(void* const* d_in, const int* in_sizes, int n_in,
                              void* d_out, int out_size, void* d_ws, size_t ws_size,
                              hipStream_t stream) {
    const float* x  = (const float*)d_in[0];
    const float* W  = (const float*)d_in[1];
    const float* b  = (const float*)d_in[2];
    const float* Wo = (const float*)d_in[3];
    const float* bo = (const float*)d_in[4];
    float* ws  = (float*)d_ws;
    float* out = (float*)d_out;

    k1_prep<<<1, 256, 0, stream>>>(x, W, b, ws);
    k2_stream<<<kNBlk, kWaves * 32, 0, stream>>>(x, ws);
    k3_final<<<1, 256, 0, stream>>>(W, b, Wo, bo, ws, out);
}
